// FJSPActorCritic_7490422964796
// MI455X (gfx1250) — matrix operand order repaired and checked by reading, not compiled, not run on hardware
//
#include <hip/hip_runtime.h>
#include <hip/hip_bf16.h>

// ---------------------------------------------------------------------------
// FJSP Actor-Critic on MI455X (gfx1250, wave32, WMMA).
//
// Actor: 262144 x (256 -> 128 -> 64 -> 1) MLP  ~= 21.5 GFLOP, tensor-core bound
//        (gather tables are L2-resident; HBM floor ~0.5us). Layers 1&2 run on
//        v_wmma_f32_16x16x32_f16 (f16 in, f32 accumulate), layer 3 on VALU +
//        lane shuffles. Weights are pre-swizzled into WMMA B-fragment order.
// Critic: deterministic 2-pass column-mean + tiny MLP (one workgroup).
// ---------------------------------------------------------------------------

typedef __attribute__((ext_vector_type(16))) _Float16 v16h;
typedef __attribute__((ext_vector_type(8)))  float    v8f;

#define N_OPS   16384
#define N_MACH  128
#define A_TOT   262144
#define H_DIM   128

// workspace byte offsets (total ~4.4 MB)
#define OFF_OPF16   0          // 16384*128 f16   = 4194304 B
#define OFF_MACHF16 4194304    //   128*128 f16   =   32768 B
#define OFF_W1F     4227072    //   256*128 f16   =   65536 B (B-frag order)
#define OFF_W2F     4292608    //   128* 64 f16   =   16384 B (B-frag order)
#define OFF_PART    4308992    //   128*128 f32   =   65536 B (critic partials)

union AFrag { v16h v; uint4 q[2]; };

// ---------------------------------------------------------------------------
// Prep: f32->f16 embedding tables + weight swizzle into WMMA B-frag layout.
// B (KxN) 16-bit layout: lane l holds column n=l&15, K = (l>>4)*8 + (j&7) + (j>>3)*16,
// j = 0..15.  frag[((nt*KT+kt)*32 + lane)*16 + j].
// ---------------------------------------------------------------------------
__global__ __launch_bounds__(256) void prep_kernel(
    const float* __restrict__ op_emb, const float* __restrict__ mach_emb,
    const float* __restrict__ aW1, const float* __restrict__ aW2,
    _Float16* __restrict__ opf16, _Float16* __restrict__ machf16,
    _Float16* __restrict__ w1f, _Float16* __restrict__ w2f) {
  int t = blockIdx.x * 256 + threadIdx.x;
  if (t < N_OPS * H_DIM) { opf16[t] = (_Float16)op_emb[t]; return; }
  t -= N_OPS * H_DIM;
  if (t < N_MACH * H_DIM) { machf16[t] = (_Float16)mach_emb[t]; return; }
  t -= N_MACH * H_DIM;
  if (t < 256 * 128) {  // W1: K=256, N=128 -> 8 ntiles x 8 ktiles
    int j = t & 15, lane = (t >> 4) & 31, kn = t >> 9;
    int kt = kn & 7, nt = kn >> 3;
    int k = kt * 32 + (lane >> 4) * 8 + (j & 7) + (j >> 3) * 16;
    int n = nt * 16 + (lane & 15);
    w1f[t] = (_Float16)aW1[k * 128 + n];
    return;
  }
  t -= 256 * 128;
  if (t < 128 * 64) {   // W2: K=128, N=64 -> 4 ntiles x 4 ktiles
    int j = t & 15, lane = (t >> 4) & 31, kn = t >> 9;
    int kt = kn & 3, nt = kn >> 2;
    int k = kt * 32 + (lane >> 4) * 8 + (j & 7) + (j >> 3) * 16;
    int n = nt * 16 + (lane & 15);
    w2f[t] = (_Float16)aW2[k * 64 + n];
  }
}

// ---------------------------------------------------------------------------
// Critic pass 1: deterministic partial column sums of op_emb (no atomics).
// ---------------------------------------------------------------------------
__global__ __launch_bounds__(256) void critic_partial_kernel(
    const float* __restrict__ op_emb, float* __restrict__ part) {
  int col = threadIdx.x & 127;
  int chunk = blockIdx.x * 2 + (threadIdx.x >> 7);   // 128 chunks of 128 rows
  const float* p = op_emb + (size_t)chunk * 128 * H_DIM + col;
  float s = 0.f;
  for (int i = 0; i < 128; ++i) s += p[i * H_DIM];
  part[chunk * 128 + col] = s;
}

// ---------------------------------------------------------------------------
// Critic pass 2: finish means, run 256->128->64->1 MLP, write value.
// ---------------------------------------------------------------------------
__global__ __launch_bounds__(256) void critic_mlp_kernel(
    const float* __restrict__ part, const float* __restrict__ mach_emb,
    const float* __restrict__ cW1, const float* __restrict__ cb1,
    const float* __restrict__ cW2, const float* __restrict__ cb2,
    const float* __restrict__ cW3, const float* __restrict__ cb3,
    float* __restrict__ outv) {
  __shared__ float gs[256], h1[128], h2[64];
  int t = threadIdx.x;
  if (t < 128) {
    float s = 0.f;
    for (int c = 0; c < 128; ++c) s += part[c * 128 + t];
    gs[t] = s * (1.f / (float)N_OPS);
  } else {
    int col = t - 128;
    float s = 0.f;
    for (int r = 0; r < N_MACH; ++r) s += mach_emb[r * H_DIM + col];
    gs[t] = s * (1.f / (float)N_MACH);
  }
  __syncthreads();
  if (t < 128) {
    float s = cb1[t];
    for (int k = 0; k < 256; ++k) s += gs[k] * cW1[k * 128 + t];
    h1[t] = fmaxf(s, 0.f);
  }
  __syncthreads();
  if (t < 64) {
    float s = cb2[t];
    for (int k = 0; k < 128; ++k) s += h1[k] * cW2[k * 64 + t];
    h2[t] = fmaxf(s, 0.f);
  }
  __syncthreads();
  if (t == 0) {
    float s = cb3[0];
    for (int k = 0; k < 64; ++k) s += h2[k] * cW3[k];
    *outv = s;
  }
}

// ---------------------------------------------------------------------------
// Actor GEMM: one wave = one 16-row M-tile per loop trip.
// Layer1: 8 ntiles x 8 ksteps WMMA f16; Layer2: 4x4 WMMA via per-wave LDS
// h1 tile; Layer3: per-lane dot + shfl_xor reduction across 16 columns.
// ---------------------------------------------------------------------------
#define NBLK 512
#define GTIL 4   // tiles per wave: 512 WG * 8 waves * 4 = 16384 tiles = 262144 rows

__global__ __launch_bounds__(256) void actor_kernel(
    const _Float16* __restrict__ opf16, const _Float16* __restrict__ machf16,
    const _Float16* __restrict__ w1f, const _Float16* __restrict__ w2f,
    const int* __restrict__ op_idx, const int* __restrict__ mach_idx,
    const unsigned char* __restrict__ valid,
    const float* __restrict__ b1, const float* __restrict__ b2,
    const float* __restrict__ w3, const float* __restrict__ b3,
    float* __restrict__ out) {
  __shared__ _Float16 s_h1[8 * 16 * H_DIM];       // 8 waves x (16x128 f16) = 32KB
  const int lane = threadIdx.x & 31;
  const int wid  = threadIdx.x >> 5;
  const int lo = lane & 15, hi = lane >> 4;
  _Float16* h1p = s_h1 + wid * (16 * H_DIM);

  float w3v[4];
#pragma unroll
  for (int nt2 = 0; nt2 < 4; ++nt2) w3v[nt2] = w3[nt2 * 16 + lo];
  const float b3v = b3[0];

  const int waveG = blockIdx.x * 8 + wid;
  for (int tt = 0; tt < GTIL; ++tt) {
    const int tile = waveG * GTIL + tt;
    const int m0 = tile * 16;
    const int row = m0 + lo;                       // this lane's action row
    const _Float16* opRow = opf16 + (size_t)op_idx[row] * H_DIM;
    const _Float16* maRow = machf16 + (size_t)mach_idx[row] * H_DIM;

    // ---- A fragments: x = concat(op_emb[oi], mach_emb[mi])  [16 x 256] f16
    // ISA 16-bit A layout: lane holds M=lo, chunks K = kt*32+hi*8 and +16.
    AFrag a[8];
#pragma unroll
    for (int kt = 0; kt < 8; ++kt) {
      const int kb0 = kt * 32 + hi * 8;
      const int kb1 = kb0 + 16;
      const _Float16* c0 = (kb0 < H_DIM) ? (opRow + kb0) : (maRow + (kb0 - H_DIM));
      const _Float16* c1 = (kb1 < H_DIM) ? (opRow + kb1) : (maRow + (kb1 - H_DIM));
      a[kt].q[0] = *(const uint4*)c0;
      a[kt].q[1] = *(const uint4*)c1;
    }

    // ---- Layer 1: [16x256] x [256x128] + b1, ReLU -> LDS h1 (row-major f16)
#pragma unroll
    for (int nt = 0; nt < 8; ++nt) {
      const float bias = b1[nt * 16 + lo];
      v8f c;
#pragma unroll
      for (int r = 0; r < 8; ++r) c[r] = bias;
#pragma unroll
      for (int kt = 0; kt < 8; ++kt) {
        AFrag bf;
        const _Float16* bp = w1f + (size_t)((nt * 8 + kt) * 32 + lane) * 16;
        bf.q[0] = *(const uint4*)bp;
        bf.q[1] = *(const uint4*)(bp + 8);
        c = __builtin_amdgcn_wmma_f32_16x16x32_f16(false, a[kt].v, false, bf.v,
                                                   (short)0, c, false, false);
      }
#pragma unroll
      for (int r = 0; r < 8; ++r) {  // C layout: col=lo, row=r+hi*8
        h1p[(r + hi * 8) * H_DIM + nt * 16 + lo] = (_Float16)fmaxf(c[r], 0.f);
      }
    }

    // ---- Layer 2 + 3: [16x128] x [128x64] + b2, ReLU, dot with w3
    float s[8];
#pragma unroll
    for (int r = 0; r < 8; ++r) s[r] = 0.f;
#pragma unroll
    for (int nt2 = 0; nt2 < 4; ++nt2) {
      const float bias = b2[nt2 * 16 + lo];
      v8f c;
#pragma unroll
      for (int r = 0; r < 8; ++r) c[r] = bias;
#pragma unroll
      for (int kt2 = 0; kt2 < 4; ++kt2) {
        AFrag a2;
        const _Float16* hp = h1p + lo * H_DIM + kt2 * 32 + hi * 8;
        a2.q[0] = *(const uint4*)hp;
        a2.q[1] = *(const uint4*)(hp + 16);
        AFrag bf;
        const _Float16* bp = w2f + (size_t)((nt2 * 4 + kt2) * 32 + lane) * 16;
        bf.q[0] = *(const uint4*)bp;
        bf.q[1] = *(const uint4*)(bp + 8);
        c = __builtin_amdgcn_wmma_f32_16x16x32_f16(false, a2.v, false, bf.v,
                                                   (short)0, c, false, false);
      }
#pragma unroll
      for (int r = 0; r < 8; ++r) s[r] += fmaxf(c[r], 0.f) * w3v[nt2];
    }

    // reduce the 16 column-lanes of each half-group
#pragma unroll
    for (int m = 1; m < 16; m <<= 1) {
#pragma unroll
      for (int r = 0; r < 8; ++r) s[r] += __shfl_xor(s[r], m, 32);
    }

    if (lo < 8) {
      float sc = 0.f;
#pragma unroll
      for (int r = 0; r < 8; ++r) if (lo == r) sc = s[r];
      const int orow = m0 + lo + hi * 8;
      out[orow] = valid[orow] ? (sc + b3v) : -1e9f;
    }
  }
}

// ---------------------------------------------------------------------------
extern "C" void kernel_launch(void* const* d_in, const int* in_sizes, int n_in,
                              void* d_out, int out_size, void* d_ws, size_t ws_size,
                              hipStream_t stream) {
  const float* op_emb   = (const float*)d_in[0];
  const float* mach_emb = (const float*)d_in[1];
  const int*   op_idx   = (const int*)d_in[2];
  const int*   mach_idx = (const int*)d_in[3];
  const unsigned char* valid = (const unsigned char*)d_in[4];  // jnp.bool_ = 1B
  const float* aW1 = (const float*)d_in[5];
  const float* ab1 = (const float*)d_in[6];
  const float* aW2 = (const float*)d_in[7];
  const float* ab2 = (const float*)d_in[8];
  const float* aW3 = (const float*)d_in[9];
  const float* ab3 = (const float*)d_in[10];
  const float* cW1 = (const float*)d_in[11];
  const float* cb1 = (const float*)d_in[12];
  const float* cW2 = (const float*)d_in[13];
  const float* cb2 = (const float*)d_in[14];
  const float* cW3 = (const float*)d_in[15];
  const float* cb3 = (const float*)d_in[16];

  char* ws = (char*)d_ws;
  _Float16* opf16   = (_Float16*)(ws + OFF_OPF16);
  _Float16* machf16 = (_Float16*)(ws + OFF_MACHF16);
  _Float16* w1f     = (_Float16*)(ws + OFF_W1F);
  _Float16* w2f     = (_Float16*)(ws + OFF_W2F);
  float*    part    = (float*)(ws + OFF_PART);

  float* logits = (float*)d_out;
  float* valueOut = logits + A_TOT;

  const int prepWork = N_OPS * H_DIM + N_MACH * H_DIM + 256 * 128 + 128 * 64;
  prep_kernel<<<(prepWork + 255) / 256, 256, 0, stream>>>(
      op_emb, mach_emb, aW1, aW2, opf16, machf16, w1f, w2f);

  critic_partial_kernel<<<64, 256, 0, stream>>>(op_emb, part);
  critic_mlp_kernel<<<1, 256, 0, stream>>>(part, mach_emb, cW1, cb1, cW2, cb2,
                                           cW3, cb3, valueOut);

  actor_kernel<<<NBLK, 256, 0, stream>>>(opf16, machf16, w1f, w2f, op_idx,
                                         mach_idx, valid, ab1, ab2, aW3, ab3,
                                         logits);
}
